// lmrk_encoder_h_8443905704051
// MI455X (gfx1250) — matrix operand order repaired and checked by reading, not compiled, not run on hardware
//
#include <hip/hip_runtime.h>
#include <hip/hip_bf16.h>
#include <math.h>

// Problem constants
#define BB    8192
#define NN    68
#define NP    80        // N padded to multiple of 16
#define FIN   2
#define FF    128
#define CC    16
#define EE    272
#define KK    256       // concat K dimension (agg | root)

typedef __attribute__((ext_vector_type(16))) _Float16 v16h;
typedef __attribute__((ext_vector_type(8)))  _Float16 v8h;
typedef __attribute__((ext_vector_type(8)))  float    v8f;

// ---------------------------------------------------------------------------
// Kernel 0: pack f16 weight matrices B[f][k] = (k<128 ? W_rel[f][k] : W_root[f][k-128]),
// build deterministic CSR of incoming edges, zero loss accumulators.
// ---------------------------------------------------------------------------
__global__ __launch_bounds__(256)
void prep_kernel(const float* __restrict__ W2_rel, const float* __restrict__ W2_root,
                 const float* __restrict__ W3_rel, const float* __restrict__ W3_root,
                 const int* __restrict__ edge_index,
                 _Float16* __restrict__ B2, _Float16* __restrict__ B3,
                 int* __restrict__ csr_off, int* __restrict__ csr_src,
                 double* __restrict__ accum)
{
    const int tid = threadIdx.x;
    for (int idx = tid; idx < FF * KK; idx += 256) {
        int f = idx >> 8, k = idx & 255;
        float w2 = (k < FF) ? W2_rel[f * FF + k] : W2_root[f * FF + (k - FF)];
        float w3 = (k < FF) ? W3_rel[f * FF + k] : W3_root[f * FF + (k - FF)];
        B2[idx] = (_Float16)w2;
        B3[idx] = (_Float16)w3;
    }

    __shared__ int cnt[NN];
    const int* src = edge_index;
    const int* dst = edge_index + EE;
    if (tid < NN) {
        int c = 0;
        for (int e = 0; e < EE; e++) if (dst[e] == tid) c++;
        cnt[tid] = c;
    }
    __syncthreads();
    if (tid == 0) {
        int run = 0;
        for (int n = 0; n < NN; n++) { csr_off[n] = run; run += cnt[n]; }
        csr_off[NN] = run;
        accum[0] = 0.0;   // link-loss sum
        accum[1] = 0.0;   // entropy sum
    }
    __syncthreads();
    if (tid < NN) {
        int w = csr_off[tid];
        for (int e = 0; e < EE; e++) if (dst[e] == tid) csr_src[w++] = src[e];
    }
}

// ---------------------------------------------------------------------------
// WMMA GEMM: C[80x128] = A[80x256] (LDS, f16) * Bmat[256x128] (global f16, [f][k]).
// One N-tile (16 cols) per wave, loop 5 M-tiles, 8 k-steps of 32.
// Fragment k order: contiguous k = kb + 16h + s, same in A and B (wave32).
// ---------------------------------------------------------------------------
static __device__ __forceinline__
void gemm_layer(const _Float16* __restrict__ A, const _Float16* __restrict__ Bmat,
                int lane, int wave, v8f acc[5])
{
    const int nlo  = lane & 15;
    const int hihf = (lane & 16) ? 1 : 0;

    // B fragments: col N = wave*16 + nlo, halves K = kb..kb+15 contiguous
    v16h bfrag[8];
    {
        const _Float16* bp = Bmat + (wave * 16 + nlo) * KK + (hihf ? 16 : 0);
        #pragma unroll
        for (int ks = 0; ks < 8; ks++)
            bfrag[ks] = *(const v16h*)(bp + ks * 32);
    }

    #pragma unroll
    for (int mt = 0; mt < 5; mt++) {
        v8f c = {};
        const _Float16* ap = A + (mt * 16 + nlo) * KK + (hihf ? 16 : 0);
        #pragma unroll
        for (int ks = 0; ks < 8; ks++) {
            v16h a = *(const v16h*)(ap + ks * 32);       // K = kb+16h .. kb+16h+15
            c = __builtin_amdgcn_wmma_f32_16x16x32_f16(
                    false, a, false, bfrag[ks], (short)0, c, false, false);
        }
        acc[mt] = c;
    }
}

// ---------------------------------------------------------------------------
// Kernel 1: one block per batch. GraphConv x3 (WMMA) + diff-pool.
// ---------------------------------------------------------------------------
__global__ __launch_bounds__(256)
void batch_kernel(const float* __restrict__ x, const float* __restrict__ adjg,
                  const float* __restrict__ s, const float* __restrict__ pos,
                  const float* __restrict__ W1_rel, const float* __restrict__ W1_root,
                  const float* __restrict__ b1, const float* __restrict__ b2,
                  const float* __restrict__ b3,
                  const _Float16* __restrict__ B2, const _Float16* __restrict__ B3,
                  const int* __restrict__ csr_off, const int* __restrict__ csr_src,
                  float* __restrict__ out, float* __restrict__ out_adj,
                  float* __restrict__ pos_out, double* __restrict__ accum)
{
    const int b    = blockIdx.x;
    const int tid  = threadIdx.x;
    const int lane = tid & 31;
    const int wave = tid >> 5;

    __shared__ __align__(16) _Float16 A[NP * KK];   // 40 KB: [agg | h] f16
    __shared__ float ssm[NN * CC];                  // softmax(s)
    __shared__ float xl[NN * FIN];
    __shared__ float aggx[NN * FIN];
    __shared__ float tmp[CC * NN];                  // s^T * adj
    __shared__ float red[256];
    __shared__ float entn[NN];
    __shared__ int   soff[NN + 1];
    __shared__ int   ssrc[EE];

    const float* xb = x    + (size_t)b * (NN * FIN);
    const float* ab = adjg + (size_t)b * (NN * NN);
    const float* sb = s    + (size_t)b * (NN * CC);

    // ---- stage inputs + CSR -------------------------------------------------
    for (int i = tid; i < NN * FIN; i += 256) xl[i] = xb[i];
    if (tid <= NN) soff[tid] = csr_off[tid];
    for (int i = tid; i < EE; i += 256) ssrc[i] = csr_src[i];
    __syncthreads();

    // ---- layer 1 (F_IN = 2, scalar) ----------------------------------------
    for (int i = tid; i < NN * FIN; i += 256) {
        int n = i >> 1, c = i & 1;
        float acc = 0.f;
        for (int e = soff[n]; e < soff[n + 1]; e++) acc += xl[ssrc[e] * FIN + c];
        aggx[i] = acc;
    }
    __syncthreads();
    for (int i = tid; i < NP * FF; i += 256) {
        int n = i >> 7, f = i & 127;
        float v = 0.f;
        if (n < NN) {
            v = aggx[n * 2] * W1_rel[f * 2] + aggx[n * 2 + 1] * W1_rel[f * 2 + 1]
              + xl[n * 2]   * W1_root[f * 2] + xl[n * 2 + 1] * W1_root[f * 2 + 1]
              + b1[f];
            v = v > 0.f ? v : 0.f;
        }
        A[n * KK + FF + f] = (_Float16)v;     // root half; pad rows -> 0
    }
    __syncthreads();

    // ---- layers 2 and 3: aggregate -> WMMA GEMM -> bias+ReLU ---------------
    #pragma unroll 1
    for (int layer = 0; layer < 2; layer++) {
        // aggregation: A[:,0:128] = sum over incoming edges of A[src,128:]
        for (int i = tid; i < NP * FF; i += 256) {
            int n = i >> 7, f = i & 127;
            float acc = 0.f;
            if (n < NN) {
                for (int e = soff[n]; e < soff[n + 1]; e++)
                    acc += (float)A[ssrc[e] * KK + FF + f];
            }
            A[n * KK + f] = (_Float16)acc;
        }
        __syncthreads();

        const _Float16* Bmat = (layer == 0) ? B2 : B3;
        const float*    bias = (layer == 0) ? b2 : b3;

        v8f acc[5];
        gemm_layer(A, Bmat, lane, wave, acc);
        __syncthreads();   // all reads of A done before overwrite

        const int col = wave * 16 + (lane & 15);
        const float bv = bias[col];
        #pragma unroll
        for (int mt = 0; mt < 5; mt++) {
            int rbase = mt * 16 + ((lane & 16) ? 8 : 0);
            #pragma unroll
            for (int v = 0; v < 8; v++) {
                float val = acc[mt][v] + bv;
                val = val > 0.f ? val : 0.f;
                A[(rbase + v) * KK + FF + col] = (_Float16)val;
            }
        }
        __syncthreads();
    }
    // h3 (f16) now lives in A[:,128:256]

    // ---- softmax(s) + entropy ----------------------------------------------
    if (tid < NN) {
        float mx = -1e30f;
        for (int c = 0; c < CC; c++) mx = fmaxf(mx, sb[tid * CC + c]);
        float e[CC]; float sum = 0.f;
        for (int c = 0; c < CC; c++) { e[c] = expf(sb[tid * CC + c] - mx); sum += e[c]; }
        float inv = 1.f / sum, ent = 0.f;
        for (int c = 0; c < CC; c++) {
            float p = e[c] * inv;
            ssm[tid * CC + c] = p;
            ent -= p * logf(p + 1e-15f);
        }
        entn[tid] = ent;
    }
    __syncthreads();

    // ---- out[b] = s^T h3 : [16 x 128] --------------------------------------
    float* outb = out + (size_t)b * (CC * FF);
    for (int i = tid; i < CC * FF; i += 256) {
        int c = i >> 7, f = i & 127;
        float acc = 0.f;
        for (int n = 0; n < NN; n++)
            acc += ssm[n * CC + c] * (float)A[n * KK + FF + f];
        outb[i] = acc;
    }

    // ---- tmp = s^T adj (adj streamed from global, read once) ---------------
    if (tid < NN) {
        const int m = tid;
        float col[CC];
        for (int c = 0; c < CC; c++) col[c] = 0.f;
        for (int n = 0; n < NN; n++) {
            float a = ab[n * NN + m];
            for (int c = 0; c < CC; c++) col[c] += ssm[n * CC + c] * a;
        }
        for (int c = 0; c < CC; c++) tmp[c * NN + m] = col[c];
    }
    __syncthreads();

    // ---- out_adj[b] = tmp * s : [16 x 16] ----------------------------------
    float* oab = out_adj + (size_t)b * (CC * CC);
    if (tid < CC * CC) {
        int c = tid >> 4, k = tid & 15;
        float acc = 0.f;
        for (int m = 0; m < NN; m++) acc += tmp[c * NN + m] * ssm[m * CC + k];
        oab[tid] = acc;
    }

    // ---- link loss partial: || adj - s s^T ||^2 (adj second streaming pass) -
    float lp = 0.f;
    for (int i = tid; i < NN * NN; i += 256) {
        int n = i / NN, m = i - n * NN;
        float d = ab[i];
        for (int c = 0; c < CC; c++) d -= ssm[n * CC + c] * ssm[m * CC + c];
        lp += d * d;
    }
    red[tid] = lp;
    __syncthreads();
    if (tid == 0) {
        float t = 0.f;
        for (int j = 0; j < 256; j++) t += red[j];     // deterministic in-block sum
        float ent = 0.f;
        for (int n = 0; n < NN; n++) ent += entn[n];
        atomicAdd(&accum[0], (double)t);
        atomicAdd(&accum[1], (double)ent);
    }

    // ---- pos passthrough ----------------------------------------------------
    const float* pb  = pos     + (size_t)b * (NN * FIN);
    float*       pob = pos_out + (size_t)b * (NN * FIN);
    for (int i = tid; i < NN * FIN; i += 256) pob[i] = pb[i];
}

// ---------------------------------------------------------------------------
// Kernel 2: finalize scalars.
// ---------------------------------------------------------------------------
__global__ void finalize_kernel(const double* __restrict__ accum,
                                float* __restrict__ link, float* __restrict__ ent)
{
    if (threadIdx.x == 0 && blockIdx.x == 0) {
        *link = (float)(sqrt(accum[0]) / ((double)BB * NN * NN));
        *ent  = (float)(accum[1] / ((double)BB * NN));
    }
}

// ---------------------------------------------------------------------------
extern "C" void kernel_launch(void* const* d_in, const int* in_sizes, int n_in,
                              void* d_out, int out_size, void* d_ws, size_t ws_size,
                              hipStream_t stream)
{
    const float* x       = (const float*)d_in[0];
    const int*   ei      = (const int*)  d_in[1];
    const float* adj     = (const float*)d_in[2];
    const float* s       = (const float*)d_in[3];
    const float* pos     = (const float*)d_in[4];
    const float* W1_rel  = (const float*)d_in[5];
    const float* W1_root = (const float*)d_in[6];
    const float* b1      = (const float*)d_in[7];
    const float* W2_rel  = (const float*)d_in[8];
    const float* W2_root = (const float*)d_in[9];
    const float* b2      = (const float*)d_in[10];
    const float* W3_rel  = (const float*)d_in[11];
    const float* W3_root = (const float*)d_in[12];
    const float* b3      = (const float*)d_in[13];

    char* ws = (char*)d_ws;
    _Float16* B2      = (_Float16*)ws;                 // 64 KB
    _Float16* B3      = (_Float16*)(ws + 65536);       // 64 KB
    int*      csr_off = (int*)(ws + 131072);           // 69 ints
    int*      csr_src = csr_off + 128;                 // 272 ints
    double*   accum   = (double*)(ws + 131072 + 2048); // 2 doubles

    float* out     = (float*)d_out;                    // [B,16,128]
    float* out_adj = out + (size_t)BB * CC * FF;       // [B,16,16]
    float* link    = out_adj + (size_t)BB * CC * CC;   // scalar
    float* ent     = link + 1;                         // scalar
    float* pos_out = ent + 1;                          // [B,68,2]

    prep_kernel<<<1, 256, 0, stream>>>(W2_rel, W2_root, W3_rel, W3_root, ei,
                                       B2, B3, csr_off, csr_src, accum);
    batch_kernel<<<BB, 256, 0, stream>>>(x, adj, s, pos, W1_rel, W1_root, b1, b2, b3,
                                         B2, B3, csr_off, csr_src,
                                         out, out_adj, pos_out, accum);
    finalize_kernel<<<1, 32, 0, stream>>>(accum, link, ent);
}
